// DSMAE_41412074668235
// MI455X (gfx1250) — matrix operand order repaired and checked by reading, not compiled, not run on hardware
//
#include <hip/hip_runtime.h>

typedef __attribute__((ext_vector_type(16))) _Float16 v16h;
typedef __attribute__((ext_vector_type(8)))  float    v8f;

// ---------------------------------------------------------------------------
// Pack weights to f16, TRANSPOSED (column-major vs the GEMM's B):
//   Wc0T[c][k], c in [0,640): c = r*128+j (rel) | 512+j (root), k in [0,64)
//   Wc1T[c][k], c in [0,320): c = r*64+j  (rel) | 256+j (root), k in [0,128)
// Column-major lets each lane's WMMA B-fragment be one contiguous 32B load.
// ---------------------------------------------------------------------------
__global__ void rgcn_pack_weights(const float* __restrict__ w0, const float* __restrict__ root0,
                                  const float* __restrict__ w1, const float* __restrict__ root1,
                                  _Float16* __restrict__ Wc0T, _Float16* __restrict__ Wc1T)
{
    int gid = blockIdx.x * blockDim.x + threadIdx.x;
    if (gid < 640 * 64) {
        int c = gid / 64, k = gid % 64;
        float v = (c < 512) ? w0[((c >> 7) * 64 + k) * 128 + (c & 127)]
                            : root0[k * 128 + (c - 512)];
        Wc0T[gid] = (_Float16)v;
    } else if (gid < 640 * 64 + 320 * 128) {
        int i = gid - 640 * 64;
        int c = i / 128, k = i % 128;
        float v = (c < 256) ? w1[((c >> 6) * 128 + k) * 64 + (c & 63)]
                            : root1[k * 64 + (c - 256)];
        Wc1T[i] = (_Float16)v;
    }
}

__global__ void rgcn_zero(int* __restrict__ cnt, float* __restrict__ stats, int ncnt)
{
    int gid = blockIdx.x * blockDim.x + threadIdx.x;
    if (gid < ncnt) cnt[gid] = 0;
    else if (gid < ncnt + 256) stats[gid - ncnt] = 0.0f;
}

__global__ void rgcn_count(const int* __restrict__ e0, const int* __restrict__ e1,
                           const int* __restrict__ e2, const int* __restrict__ e3,
                           int* __restrict__ cnt, int E, int N)
{
    long long gid = (long long)blockIdx.x * blockDim.x + threadIdx.x;
    if (gid >= 4LL * E) return;
    int r = (int)(gid / E), e = (int)(gid % E);
    const int* ed = (r == 0) ? e0 : (r == 1) ? e1 : (r == 2) ? e2 : e3;
    atomicAdd(&cnt[(size_t)r * N + ed[E + e]], 1);
}

__global__ void rgcn_inv(const int* __restrict__ cnt, float* __restrict__ inv, int n)
{
    int gid = blockIdx.x * blockDim.x + threadIdx.x;
    if (gid < n) inv[gid] = 1.0f / fmaxf((float)cnt[gid], 1.0f);
}

// ---------------------------------------------------------------------------
// WMMA GEMM: C[f16, M x Ncols] = A[f32, M x K] * BT[f16, Ncols x K]
// One wave per 16x16 tile; K stepped by 32 via v_wmma_f32_16x16x32_f16.
// A-frag: lane(l<16)=row l: h[0..15]=K0..15; l>=16: K16..31
//         -> two aligned 32B v8f loads per lane, cvt to f16.
// B-frag: col = lane%16; lanes 0-15: K=k0+j, lanes 16-31: K=k0+16+j
//         -> one aligned 32B v16h load per lane from column-major BT.
// C/D:    col = lane%16, c[i] = row i + 8*(lane/16)
// ---------------------------------------------------------------------------
__global__ __launch_bounds__(128)
void rgcn_gemm_wmma(const float* __restrict__ A, const _Float16* __restrict__ BT,
                    _Float16* __restrict__ C, int K, int Ncols, int Ntiles, int totalTiles)
{
    int tile = blockIdx.x * 4 + threadIdx.y;       // 4 waves per block
    if (tile >= totalTiles) return;                // uniform per wave -> EXEC all-1s inside
    int lane = threadIdx.x;
    int hb = lane >> 4;                            // half-wave select
    int mn = lane & 15;
    int m0 = (tile / Ntiles) << 4;
    int n0 = (tile % Ntiles) << 4;

    const float*    arow = A  + (size_t)(m0 + mn) * K + 16 * hb;
    const _Float16* bcol = BT + (size_t)(n0 + mn) * K + 16 * hb;

    v8f acc = {0.f, 0.f, 0.f, 0.f, 0.f, 0.f, 0.f, 0.f};
    for (int k0 = 0; k0 < K; k0 += 32) {
        v8f a0 = *(const v8f*)(arow + k0);          // K = k0+16*hb  .. +8
        v8f a1 = *(const v8f*)(arow + k0 + 8);      // K = k0+16*hb+8.. +8
        v16h a;
        #pragma unroll
        for (int j = 0; j < 8; ++j) {
            a[j]     = (_Float16)a0[j];
            a[j + 8] = (_Float16)a1[j];
        }
        v16h b = *(const v16h*)(bcol + k0);         // 16 contiguous f16 of this column
        acc = __builtin_amdgcn_wmma_f32_16x16x32_f16(
            /*neg_a=*/false, a, /*neg_b=*/false, b,
            /*c_mod=*/(short)0, acc, /*reuse_a=*/false, /*reuse_b=*/false);
    }
    _Float16* cp = C + (size_t)(m0 + 8 * hb) * Ncols + n0 + mn;
    #pragma unroll
    for (int i = 0; i < 8; ++i) cp[(size_t)i * Ncols] = (_Float16)acc[i];
}

// out[n,j] = T[n, rootOff+j] + bias[j]
__global__ void rgcn_init_out(const _Float16* __restrict__ T, int tStride, int rootOff,
                              const float* __restrict__ bias, float* __restrict__ out,
                              int cols, long long total)
{
    long long gid = (long long)blockIdx.x * blockDim.x + threadIdx.x;
    if (gid >= total) return;
    long long n = gid / cols;
    int j = (int)(gid % cols);
    out[gid] = (float)T[n * tStride + rootOff + j] + bias[j];
}

// One wave per edge; each lane adds outCols/32 columns: out[dst] += T[src,rel]*inv[r,dst]
__global__ void rgcn_scatter(const int* __restrict__ e0, const int* __restrict__ e1,
                             const int* __restrict__ e2, const int* __restrict__ e3,
                             const float* __restrict__ inv, const _Float16* __restrict__ T,
                             int tStride, float* __restrict__ out, int outCols, int E, int N)
{
    long long gid = (long long)blockIdx.x * blockDim.x + threadIdx.x;
    int lane = (int)(gid & 31);
    long long eg = gid >> 5;
    if (eg >= 4LL * E) return;
    int r = (int)(eg / E), e = (int)(eg % E);
    const int* ed = (r == 0) ? e0 : (r == 1) ? e1 : (r == 2) ? e2 : e3;
    int src = ed[e], dst = ed[E + e];
    float w = inv[(size_t)r * N + dst];
    int cpl = outCols >> 5;
    const _Float16* t = T + (size_t)src * tStride + r * outCols + lane * cpl;
    float* o = out + (size_t)dst * outCols + lane * cpl;
    for (int c = 0; c < cpl; ++c) atomicAdd(o + c, (float)t[c] * w);
}

// stats[0..127]=sum, stats[128..255]=sumsq over rows (channel = tid&127)
__global__ void rgcn_bn_stats(const float* __restrict__ x, float* __restrict__ stats, int N)
{
    __shared__ float sh[512];
    int j = threadIdx.x & 127;
    int g = threadIdx.x >> 7;   // 0..1 row groups
    float s = 0.f, s2 = 0.f;
    for (int n = blockIdx.x * 2 + g; n < N; n += gridDim.x * 2) {
        float v = x[(size_t)n * 128 + j];
        s += v; s2 += v * v;
    }
    sh[threadIdx.x] = s;
    sh[256 + threadIdx.x] = s2;
    __syncthreads();
    if (g == 0) {
        atomicAdd(&stats[j],       sh[threadIdx.x] + sh[threadIdx.x + 128]);
        atomicAdd(&stats[128 + j], sh[256 + threadIdx.x] + sh[256 + threadIdx.x + 128]);
    }
}

__global__ void rgcn_bn_elu(const float* __restrict__ x, const float* __restrict__ stats,
                            const float* __restrict__ gamma, const float* __restrict__ beta,
                            float* __restrict__ h, int N)
{
    long long gid = (long long)blockIdx.x * blockDim.x + threadIdx.x;
    if (gid >= (long long)N * 128) return;
    int j = (int)(gid & 127);
    float invN = 1.0f / (float)N;
    float mu  = stats[j] * invN;
    float var = stats[128 + j] * invN - mu * mu;
    float y = gamma[j] * (x[gid] - mu) * rsqrtf(var + 1e-5f) + beta[j];
    h[gid] = (y > 0.f) ? y : expm1f(y);
}

extern "C" void kernel_launch(void* const* d_in, const int* in_sizes, int n_in,
                              void* d_out, int out_size, void* d_ws, size_t ws_size,
                              hipStream_t stream)
{
    const int* e0 = (const int*)d_in[0];
    const int* e1 = (const int*)d_in[1];
    const int* e2 = (const int*)d_in[2];
    const int* e3 = (const int*)d_in[3];
    const float* emb   = (const float*)d_in[4];
    const float* w0    = (const float*)d_in[5];
    const float* root0 = (const float*)d_in[6];
    const float* b0    = (const float*)d_in[7];
    const float* gamma = (const float*)d_in[8];
    const float* beta  = (const float*)d_in[9];
    const float* w1    = (const float*)d_in[10];
    const float* root1 = (const float*)d_in[11];
    const float* b1    = (const float*)d_in[12];
    float* out = (float*)d_out;

    const int E = in_sizes[0] / 2;      // 800000
    const int N = in_sizes[4] / 64;     // 100000

    // -------- workspace carve-out (256B aligned), ~234 MB total -------------
    char* base = (char*)d_ws;
    size_t off = 0;
    auto take = [&](size_t bytes) -> char* {
        char* p = base + off;
        off += (bytes + 255) & ~(size_t)255;
        return p;
    };
    int*      cnt   = (int*)     take((size_t)4 * N * sizeof(int));
    float*    inv   = (float*)   take((size_t)4 * N * sizeof(float));
    float*    stats = (float*)   take(256 * sizeof(float));
    _Float16* Wc0T  = (_Float16*)take((size_t)640 * 64 * sizeof(_Float16));
    _Float16* Wc1T  = (_Float16*)take((size_t)320 * 128 * sizeof(_Float16));
    float*    out0  = (float*)   take((size_t)N * 128 * sizeof(float));
    float*    h1    = (float*)   take((size_t)N * 128 * sizeof(float));
    _Float16* T     = (_Float16*)take((size_t)N * 640 * sizeof(_Float16)); // T0; reused as T1

    // -------- pipeline ------------------------------------------------------
    rgcn_pack_weights<<<(640*64 + 320*128 + 255) / 256, 256, 0, stream>>>(
        w0, root0, w1, root1, Wc0T, Wc1T);

    int nz = 4 * N + 256;
    rgcn_zero<<<(nz + 255) / 256, 256, 0, stream>>>(cnt, stats, 4 * N);

    long long ce = 4LL * E;
    rgcn_count<<<(int)((ce + 255) / 256), 256, 0, stream>>>(e0, e1, e2, e3, cnt, E, N);
    rgcn_inv<<<(4 * N + 255) / 256, 256, 0, stream>>>(cnt, inv, 4 * N);

    // Layer 0: T0[N,640] = emb[N,64] @ Wc0[64,640]   (cols 512..639 = root)
    int tiles0 = (N / 16) * (640 / 16);
    rgcn_gemm_wmma<<<(tiles0 + 3) / 4, dim3(32, 4), 0, stream>>>(
        emb, Wc0T, T, 64, 640, 640 / 16, tiles0);

    long long t0 = (long long)N * 128;
    rgcn_init_out<<<(int)((t0 + 255) / 256), 256, 0, stream>>>(T, 640, 512, b0, out0, 128, t0);

    long long sthreads = 4LL * E * 32;
    rgcn_scatter<<<(int)((sthreads + 255) / 256), 256, 0, stream>>>(
        e0, e1, e2, e3, inv, T, 640, out0, 128, E, N);

    rgcn_bn_stats<<<512, 256, 0, stream>>>(out0, stats, N);
    rgcn_bn_elu<<<(int)((t0 + 255) / 256), 256, 0, stream>>>(out0, stats, gamma, beta, h1, N);

    // Layer 1: T1[N,320] = h1[N,128] @ Wc1[128,320]  (cols 256..319 = root)
    int tiles1 = (N / 16) * (320 / 16);
    rgcn_gemm_wmma<<<(tiles1 + 3) / 4, dim3(32, 4), 0, stream>>>(
        h1, Wc1T, T, 128, 320, 320 / 16, tiles1);

    long long t1 = (long long)N * 64;
    rgcn_init_out<<<(int)((t1 + 255) / 256), 256, 0, stream>>>(T, 320, 256, b1, out, 64, t1);

    rgcn_scatter<<<(int)((sthreads + 255) / 256), 256, 0, stream>>>(
        e0, e1, e2, e3, inv, T, 320, out, 64, E, N);
}
